// ESMAttentionBlock_91122026152391
// MI455X (gfx1250) — matrix operand order repaired and checked by reading, not compiled, not run on hardware
//
#include <hip/hip_runtime.h>
#include <hip/hip_bf16.h>
#include <math.h>

typedef __attribute__((ext_vector_type(8)))  __bf16 v8bf;
typedef __attribute__((ext_vector_type(16))) __bf16 v16bf;
typedef __attribute__((ext_vector_type(8)))  float  v8f;

#define Bsz 2
#define Ssz 2048
#define Esz 768
#define Hsz 12
#define Kdim 64
#define Fsz 3072
#define Msz (Bsz*Ssz)   /* 4096 rows */

#define ASTR 40   /* LDS stride (halves) for 32-wide K tiles: 80B, 16B-aligned runs */
#define QSTR 72   /* LDS stride (halves) for 64-wide K tiles: 144B, 16B-aligned runs */

// ---- CDNA5 async global->LDS staging (ASYNCcnt) ----
__device__ __forceinline__ unsigned lds_off_of(const void* p){
    return (unsigned)(unsigned long long)(uintptr_t)p;   // LDS aperture: low 32 bits = LDS offset
}
__device__ __forceinline__ void async_b128(const void* lds_dst, const void* gsrc){
    asm volatile("global_load_async_to_lds_b128 %0, %1, off"
                 :: "v"(lds_off_of(lds_dst)), "v"((unsigned long long)(uintptr_t)gsrc)
                 : "memory");
}
__device__ __forceinline__ void wait_async(){
    asm volatile("s_wait_asynccnt 0x0" ::: "memory");
}
__device__ __forceinline__ v16bf frag_cat(v8bf a, v8bf b){
    return __builtin_shufflevector(a, b, 0,1,2,3,4,5,6,7,8,9,10,11,12,13,14,15);
}

__device__ __forceinline__ float gelu_tanh(float x){
    float x3 = x*x*x;
    return 0.5f*x*(1.0f + tanhf(0.7978845608028654f*(x + 0.044715f*x3)));
}

// ---------- f32 [K][N] -> bf16 transposed [N][K] (one-time weight prep) ----------
__global__ void cvtT_kernel(const float* __restrict__ in, __bf16* __restrict__ out, int K, int N){
    int i = blockIdx.x*blockDim.x + threadIdx.x;
    if (i < K*N){
        int k = i / N, n = i % N;
        out[(size_t)n*K + k] = (__bf16)in[i];
    }
}

// ---------- LayerNorm over rows of Esz, bf16 out ----------
__global__ void ln_kernel(const float* __restrict__ x, const float* __restrict__ scale,
                          const float* __restrict__ offset, __bf16* __restrict__ out){
    __shared__ float red[256];
    int row = blockIdx.x;
    const float* xr = x + (size_t)row*Esz;
    int tid = threadIdx.x;
    float s = 0.f;
    for (int j = tid; j < Esz; j += 256) s += xr[j];
    red[tid] = s; __syncthreads();
    for (int off = 128; off > 0; off >>= 1){ if (tid < off) red[tid] += red[tid+off]; __syncthreads(); }
    float mu = red[0] * (1.0f/Esz);
    __syncthreads();
    float s2 = 0.f;
    for (int j = tid; j < Esz; j += 256){ float d = xr[j]-mu; s2 += d*d; }
    red[tid] = s2; __syncthreads();
    for (int off = 128; off > 0; off >>= 1){ if (tid < off) red[tid] += red[tid+off]; __syncthreads(); }
    float inv = rsqrtf(red[0]*(1.0f/Esz) + 1e-5f);
    for (int j = tid; j < Esz; j += 256)
        out[(size_t)row*Esz + j] = (__bf16)(scale[j]*(xr[j]-mu)*inv + offset[j]);
}

// ---------- generic WMMA bf16 GEMM: out = epilogue(A[M,K] * Wt[N,K]^T + bias) ----------
// Both tiles async-staged row-contiguous, double-buffered LDS ping-pong.
__global__ void gemm_bf16_kernel(const __bf16* __restrict__ A, const __bf16* __restrict__ Wt,
                                 const float* __restrict__ bias, const float* __restrict__ resid,
                                 float* __restrict__ outf, __bf16* __restrict__ outb,
                                 int M, int N, int Kd, int apply_gelu){
    __shared__ __bf16 As[2][128*ASTR];   // [row][k], k<32
    __shared__ __bf16 Bs[2][128*ASTR];   // [col][k]
    int tid  = threadIdx.x;
    int lane = tid & 31, wid = tid >> 5;
    int hi   = lane >> 4, ln16 = lane & 15;
    int wrow = wid >> 1, wcol = wid & 1;
    int m0 = blockIdx.y*128, n0 = blockIdx.x*128;

    int cr = (tid*2) >> 2;            // 2 chunks/thread: rows/cols 0..127
    int cc0 = ((tid*2) & 3)*8;        // chunk half-offsets within 32-wide k
    auto stage = [&](int buf, int k0){
        #pragma unroll
        for (int i = 0; i < 2; ++i){
            int ch = cc0 + i*8;       // (tid*2+i) has same >>2 when (tid*2)&3 in {0,2}; recompute safely below
            int cid = tid*2 + i;
            int r  = cid >> 2;
            int c  = (cid & 3)*8;
            async_b128(&As[buf][r*ASTR + c], &A [(size_t)(m0 + r)*Kd + k0 + c]);
            async_b128(&Bs[buf][r*ASTR + c], &Wt[(size_t)(n0 + r)*Kd + k0 + c]);
            (void)ch; (void)cr;
        }
    };

    v8f acc[2][4] = {};
    stage(0, 0);
    int buf = 0;
    for (int k0 = 0; k0 < Kd; k0 += 32){
        wait_async();
        __syncthreads();
        if (k0 + 32 < Kd) stage(buf ^ 1, k0 + 32);
        #pragma unroll
        for (int tr = 0; tr < 2; ++tr){
            const v8bf* ap = (const v8bf*)&As[buf][(wrow*32 + tr*16 + ln16)*ASTR + hi*8];
            v16bf af = frag_cat(ap[0], ap[2]);    // k runs: [hi*8,+8) and [hi*8+16,+8)
            #pragma unroll
            for (int tc = 0; tc < 4; ++tc){
                const v8bf* bp = (const v8bf*)&Bs[buf][(wcol*64 + tc*16 + ln16)*ASTR + hi*8];
                v16bf bfv = frag_cat(bp[0], bp[2]);
                acc[tr][tc] = __builtin_amdgcn_wmma_f32_16x16x32_bf16(
                    false, af, false, bfv, (short)0, acc[tr][tc], false, false);
            }
        }
        buf ^= 1;
    }
    #pragma unroll
    for (int tr = 0; tr < 2; ++tr)
    #pragma unroll
    for (int tc = 0; tc < 4; ++tc){
        int gcol = n0 + wcol*64 + tc*16 + ln16;
        float bv = bias[gcol];
        #pragma unroll
        for (int r = 0; r < 8; ++r){
            int grow = m0 + wrow*32 + tr*16 + hi*8 + r;
            float val = acc[tr][tc][r] + bv;
            if (apply_gelu) val = gelu_tanh(val);
            size_t o = (size_t)grow*N + gcol;
            if (resid) val += resid[o];
            if (outf)  outf[o] = val;
            if (outb)  outb[o] = (__bf16)val;
        }
    }
}

// ---------- RoPE: f32 q/k -> bf16 (q scaled by 1/sqrt(64)) ----------
__global__ void rope_kernel(const float* __restrict__ qf, const float* __restrict__ kf,
                            __bf16* __restrict__ qb, __bf16* __restrict__ kb){
    int idx = blockIdx.x*blockDim.x + threadIdx.x;   // over B*S*H*32
    if (idx >= Bsz*Ssz*Hsz*32) return;
    int d = idx & 31;
    int h = (idx >> 5) % Hsz;
    int s = ((idx >> 5) / Hsz) % Ssz;
    int b = idx / (32*Hsz*Ssz);
    size_t base = ((size_t)(b*Ssz + s))*Esz + h*Kdim + d;
    float fr = (float)s * expf(-(float)d * (9.210340371976184f/32.0f));
    float c = cosf(fr), sn = sinf(fr);
    float q1 = qf[base], q2 = qf[base+32];
    qb[base]    = (__bf16)(0.125f*(q1*c - q2*sn));
    qb[base+32] = (__bf16)(0.125f*(q2*c + q1*sn));
    float k1 = kf[base], k2 = kf[base+32];
    kb[base]    = (__bf16)(k1*c - k2*sn);
    kb[base+32] = (__bf16)(k2*c + k1*sn);
}

// ---------- v [b,t,h,d] -> vt [b,h,d,t] (bf16, one-time) ----------
__global__ void transpose_v_kernel(const __bf16* __restrict__ vb, __bf16* __restrict__ vt){
    int idx = blockIdx.x*blockDim.x + threadIdx.x;   // over B*S*H*K
    if (idx >= Bsz*Ssz*Hsz*Kdim) return;
    int d = idx & (Kdim-1);
    int h = (idx >> 6) % Hsz;
    int t = ((idx >> 6) / Hsz) % Ssz;
    int b = idx / (Kdim*Hsz*Ssz);
    vt[(((size_t)(b*Hsz + h))*Kdim + d)*Ssz + t] = vb[idx];
}

// ---------- logits = (q*0.125) . k, masked, written raw to attn_w ----------
__global__ void logits_kernel(const __bf16* __restrict__ qb, const __bf16* __restrict__ kb,
                              const unsigned char* __restrict__ mask, float* __restrict__ attnw){
    __shared__ __bf16 qs[128*QSTR];
    __shared__ __bf16 ks[128*QSTR];
    int tid  = threadIdx.x;
    int lane = tid & 31, wid = tid >> 5;
    int hi   = lane >> 4, ln16 = lane & 15;
    int wrow = wid >> 1, wcol = wid & 1;
    int bh = blockIdx.z; int b = bh / Hsz, h = bh % Hsz;
    int t0 = blockIdx.y*128, T0 = blockIdx.x*128;
    #pragma unroll
    for (int i = 0; i < 4; ++i){                 // 1024 16B chunks per tile, 4 per thread
        int cid = tid*4 + i;
        int r = cid >> 3, ch = (cid & 7)*8;
        async_b128(&qs[r*QSTR + ch], &qb[((size_t)(b*Ssz + t0 + r))*Esz + h*Kdim + ch]);
        async_b128(&ks[r*QSTR + ch], &kb[((size_t)(b*Ssz + T0 + r))*Esz + h*Kdim + ch]);
    }
    wait_async();
    __syncthreads();
    v8f acc[2][4] = {};
    #pragma unroll
    for (int k0 = 0; k0 < Kdim; k0 += 32){
        #pragma unroll
        for (int tr = 0; tr < 2; ++tr){
            const v8bf* ap = (const v8bf*)&qs[(wrow*32 + tr*16 + ln16)*QSTR + k0 + hi*8];
            v16bf af = frag_cat(ap[0], ap[2]);
            #pragma unroll
            for (int tc = 0; tc < 4; ++tc){
                const v8bf* bp = (const v8bf*)&ks[(wcol*64 + tc*16 + ln16)*QSTR + k0 + hi*8];
                v16bf bfv = frag_cat(bp[0], bp[2]);
                acc[tr][tc] = __builtin_amdgcn_wmma_f32_16x16x32_bf16(
                    false, af, false, bfv, (short)0, acc[tr][tc], false, false);
            }
        }
    }
    #pragma unroll
    for (int tr = 0; tr < 2; ++tr)
    #pragma unroll
    for (int tc = 0; tc < 4; ++tc){
        int gT = T0 + wcol*64 + tc*16 + ln16;
        #pragma unroll
        for (int r = 0; r < 8; ++r){
            int gt = t0 + wrow*32 + tr*16 + hi*8 + r;
            float val = acc[tr][tc][r];
            if (!mask[((size_t)b*Ssz + gt)*Ssz + gT]) val = -1e30f;
            attnw[(((size_t)bh)*Ssz + gt)*Ssz + gT] = val;
        }
    }
}

// ---------- row softmax over 2048 (in place) ----------
__global__ void softmax_kernel(float* __restrict__ attnw){
    __shared__ float red[256];
    float* p = attnw + (size_t)blockIdx.x*Ssz;
    int tid = threadIdx.x;
    float v[8];
    float mx = -3.4e38f;
    #pragma unroll
    for (int j = 0; j < 8; ++j){ v[j] = p[tid + j*256]; mx = fmaxf(mx, v[j]); }
    red[tid] = mx; __syncthreads();
    for (int off = 128; off > 0; off >>= 1){ if (tid < off) red[tid] = fmaxf(red[tid], red[tid+off]); __syncthreads(); }
    mx = red[0]; __syncthreads();
    float s = 0.f;
    #pragma unroll
    for (int j = 0; j < 8; ++j){ v[j] = expf(v[j]-mx); s += v[j]; }
    red[tid] = s; __syncthreads();
    for (int off = 128; off > 0; off >>= 1){ if (tid < off) red[tid] += red[tid+off]; __syncthreads(); }
    float inv = 1.0f/red[0];
    #pragma unroll
    for (int j = 0; j < 8; ++j) p[tid + j*256] = v[j]*inv;
}

// ---------- ctx = attn_w @ v : double-buffered; V async from vt, A cvt f32->bf16 ----------
__global__ void av_kernel(const float* __restrict__ attnw, const __bf16* __restrict__ vt,
                          __bf16* __restrict__ ctx){
    __shared__ __bf16 As[2][128*ASTR];   // [t][T-sub], probabilities as bf16
    __shared__ __bf16 Vs[2][64*ASTR];    // [d][T-sub]
    int tid  = threadIdx.x;
    int lane = tid & 31, wid = tid >> 5;
    int hi   = lane >> 4, ln16 = lane & 15;
    int wrow = wid >> 1, wcol = wid & 1;
    int bh = blockIdx.y; int b = bh / Hsz, h = bh % Hsz;
    int t0 = blockIdx.x*128;

    auto stage_v = [&](int buf, int k0){
        int d  = tid >> 2;            // 256 chunks: 64 d-rows x 4 chunks of 8 T
        int ch = (tid & 3)*8;
        async_b128(&Vs[buf][d*ASTR + ch], &vt[(((size_t)bh)*Kdim + d)*Ssz + k0 + ch]);
    };
    auto stage_a = [&](int buf, int k0){
        int r  = tid >> 1;            // 128 rows x 2 chunks of 16
        int c0 = (tid & 1)*16;
        const float4* ap4 = (const float4*)&attnw[(((size_t)bh)*Ssz + t0 + r)*Ssz + k0 + c0];
        float4 f0 = ap4[0], f1 = ap4[1], f2 = ap4[2], f3 = ap4[3];
        v8bf o0, o1;
        o0[0]=(__bf16)f0.x; o0[1]=(__bf16)f0.y; o0[2]=(__bf16)f0.z; o0[3]=(__bf16)f0.w;
        o0[4]=(__bf16)f1.x; o0[5]=(__bf16)f1.y; o0[6]=(__bf16)f1.z; o0[7]=(__bf16)f1.w;
        o1[0]=(__bf16)f2.x; o1[1]=(__bf16)f2.y; o1[2]=(__bf16)f2.z; o1[3]=(__bf16)f2.w;
        o1[4]=(__bf16)f3.x; o1[5]=(__bf16)f3.y; o1[6]=(__bf16)f3.z; o1[7]=(__bf16)f3.w;
        *(v8bf*)&As[buf][r*ASTR + c0]     = o0;
        *(v8bf*)&As[buf][r*ASTR + c0 + 8] = o1;
    };

    v8f acc[2][2] = {};
    stage_v(0, 0);
    stage_a(0, 0);
    int buf = 0;
    for (int k0 = 0; k0 < Ssz; k0 += 32){
        wait_async();
        __syncthreads();
        if (k0 + 32 < Ssz){
            stage_v(buf ^ 1, k0 + 32);
            stage_a(buf ^ 1, k0 + 32);
        }
        #pragma unroll
        for (int tr = 0; tr < 2; ++tr){
            const v8bf* ap = (const v8bf*)&As[buf][(wrow*32 + tr*16 + ln16)*ASTR + hi*8];
            v16bf af = frag_cat(ap[0], ap[2]);
            #pragma unroll
            for (int tc = 0; tc < 2; ++tc){
                const v8bf* bp = (const v8bf*)&Vs[buf][(wcol*32 + tc*16 + ln16)*ASTR + hi*8];
                v16bf bfv = frag_cat(bp[0], bp[2]);
                acc[tr][tc] = __builtin_amdgcn_wmma_f32_16x16x32_bf16(
                    false, af, false, bfv, (short)0, acc[tr][tc], false, false);
            }
        }
        buf ^= 1;
    }
    #pragma unroll
    for (int tr = 0; tr < 2; ++tr)
    #pragma unroll
    for (int tc = 0; tc < 2; ++tc){
        int gd = wcol*32 + tc*16 + ln16;
        #pragma unroll
        for (int r = 0; r < 8; ++r){
            int gt = t0 + wrow*32 + tr*16 + hi*8 + r;
            ctx[((size_t)(b*Ssz + gt))*Esz + h*Kdim + gd] = (__bf16)acc[tr][tc][r];
        }
    }
}

extern "C" void kernel_launch(void* const* d_in, const int* in_sizes, int n_in,
                              void* d_out, int out_size, void* d_ws, size_t ws_size,
                              hipStream_t stream){
    const float* x    = (const float*)d_in[0];
    const unsigned char* mask = (const unsigned char*)d_in[1];
    const float* wq   = (const float*)d_in[2];
    const float* bq   = (const float*)d_in[3];
    const float* wk   = (const float*)d_in[4];
    const float* bk   = (const float*)d_in[5];
    const float* wv   = (const float*)d_in[6];
    const float* bv   = (const float*)d_in[7];
    const float* wo   = (const float*)d_in[8];
    const float* bo   = (const float*)d_in[9];
    const float* ln1s = (const float*)d_in[10];
    const float* ln1o = (const float*)d_in[11];
    const float* ln2s = (const float*)d_in[12];
    const float* ln2o = (const float*)d_in[13];
    const float* fc1w = (const float*)d_in[14];
    const float* fc1b = (const float*)d_in[15];
    const float* fc2w = (const float*)d_in[16];
    const float* fc2b = (const float*)d_in[17];

    float* xout  = (float*)d_out;
    float* attnw = xout + (size_t)Msz*Esz;   // outputs: (x, attn_w) concatenated

    char* p = (char*)d_ws;
    auto alloc = [&](size_t bytes)->void*{ void* r = (void*)p; p += (bytes + 255) & ~(size_t)255; return r; };
    __bf16* wqbT   = (__bf16*)alloc((size_t)Esz*Esz*2);   // [N][K] transposed bf16
    __bf16* wkbT   = (__bf16*)alloc((size_t)Esz*Esz*2);
    __bf16* wvbT   = (__bf16*)alloc((size_t)Esz*Esz*2);
    __bf16* wobT   = (__bf16*)alloc((size_t)Esz*Esz*2);
    __bf16* fc1wbT = (__bf16*)alloc((size_t)Esz*Fsz*2);   // [F][E]
    __bf16* fc2wbT = (__bf16*)alloc((size_t)Fsz*Esz*2);   // [E][F]
    __bf16* hbf   = (__bf16*)alloc((size_t)Msz*Esz*2);
    float*  qf    = (float*) alloc((size_t)Msz*Esz*4);
    float*  kf    = (float*) alloc((size_t)Msz*Esz*4);
    __bf16* qbf   = (__bf16*)alloc((size_t)Msz*Esz*2);
    __bf16* kbf   = (__bf16*)alloc((size_t)Msz*Esz*2);
    __bf16* vbf   = (__bf16*)alloc((size_t)Msz*Esz*2);
    __bf16* vtbf  = (__bf16*)alloc((size_t)Msz*Esz*2);    // [b,h,d,t]
    __bf16* ctxbf = (__bf16*)alloc((size_t)Msz*Esz*2);
    float*  x1    = (float*) alloc((size_t)Msz*Esz*4);
    __bf16* h2bf  = (__bf16*)alloc((size_t)Msz*Esz*2);
    __bf16* a1bf  = (__bf16*)alloc((size_t)Msz*Fsz*2);

    int ne = Esz*Esz, nf = Esz*Fsz;
    cvtT_kernel<<<(ne+255)/256, 256, 0, stream>>>(wq,   wqbT,   Esz, Esz);
    cvtT_kernel<<<(ne+255)/256, 256, 0, stream>>>(wk,   wkbT,   Esz, Esz);
    cvtT_kernel<<<(ne+255)/256, 256, 0, stream>>>(wv,   wvbT,   Esz, Esz);
    cvtT_kernel<<<(ne+255)/256, 256, 0, stream>>>(wo,   wobT,   Esz, Esz);
    cvtT_kernel<<<(nf+255)/256, 256, 0, stream>>>(fc1w, fc1wbT, Esz, Fsz);
    cvtT_kernel<<<(nf+255)/256, 256, 0, stream>>>(fc2w, fc2wbT, Fsz, Esz);

    ln_kernel<<<Msz, 256, 0, stream>>>(x, ln1s, ln1o, hbf);

    dim3 gE(Esz/128, Msz/128);   // (6, 32)
    dim3 gF(Fsz/128, Msz/128);   // (24, 32)

    gemm_bf16_kernel<<<gE, 256, 0, stream>>>(hbf, wqbT, bq, nullptr, qf, nullptr, Msz, Esz, Esz, 0);
    gemm_bf16_kernel<<<gE, 256, 0, stream>>>(hbf, wkbT, bk, nullptr, kf, nullptr, Msz, Esz, Esz, 0);
    gemm_bf16_kernel<<<gE, 256, 0, stream>>>(hbf, wvbT, bv, nullptr, nullptr, vbf, Msz, Esz, Esz, 0);

    rope_kernel<<<(Bsz*Ssz*Hsz*32)/256, 256, 0, stream>>>(qf, kf, qbf, kbf);
    transpose_v_kernel<<<(Bsz*Ssz*Hsz*Kdim)/256, 256, 0, stream>>>(vbf, vtbf);

    dim3 gl(Ssz/128, Ssz/128, Bsz*Hsz);
    logits_kernel<<<gl, 256, 0, stream>>>(qbf, kbf, mask, attnw);

    softmax_kernel<<<Bsz*Hsz*Ssz, 256, 0, stream>>>(attnw);

    dim3 gav(Ssz/128, Bsz*Hsz);
    av_kernel<<<gav, 256, 0, stream>>>(attnw, vtbf, ctxbf);

    gemm_bf16_kernel<<<gE, 256, 0, stream>>>(ctxbf, wobT, bo, x, x1, nullptr, Msz, Esz, Esz, 0);

    ln_kernel<<<Msz, 256, 0, stream>>>(x1, ln2s, ln2o, h2bf);

    gemm_bf16_kernel<<<gF, 256, 0, stream>>>(h2bf, fc1wbT, fc1b, nullptr, nullptr, a1bf, Msz, Fsz, Esz, 1);

    gemm_bf16_kernel<<<gE, 256, 0, stream>>>(a1bf, fc2wbT, fc2b, x1, xout, nullptr, Msz, Esz, Fsz, 0);
}
